// BreakthroughSNN_89163521065770
// MI455X (gfx1250) — matrix operand order repaired and checked by reading, not compiled, not run on hardware
//
#include <hip/hip_runtime.h>
#include <hip/hip_bf16.h>

// ---------------- model constants ----------------
#define B_      4
#define S_      512
#define T_      16
#define DM_     512
#define DS_     64
#define V_      32000
#define H_      4
#define DH_     128          // DM_/H_
#define DECAY_  0.6065306597126334f   // exp(-1/TAU), TAU=2
#define ADAPT_  0.1f
#define MAXLAT_ 10.0f

typedef __bf16 bf16_t;
typedef __attribute__((ext_vector_type(16))) __bf16 v16bf;
typedef __attribute__((ext_vector_type(8)))  float  v8f;
typedef __attribute__((ext_vector_type(2)))  float  v2f;
typedef __attribute__((ext_vector_type(4)))  unsigned int u32x4;
typedef __attribute__((ext_vector_type(8)))  int          i32x8;
typedef __attribute__((ext_vector_type(4)))  int          i32x4;

// ---------------- WMMA fragment helpers (wave32 layouts per CDNA5 ISA §7.12.2) ----
// A (16x32 bf16): lane m=lane&15; elems 0..7 -> K=kb..kb+7, elems 8..15 -> K=kb+16..kb+23, kb=(lane<16?0:8)
__device__ __forceinline__ v16bf ld_a16(const bf16_t* row, int lane) {
  const int kb = (lane & 16) ? 8 : 0;
  v16bf f;
#pragma unroll
  for (int i = 0; i < 8; ++i) { f[i] = row[kb + i]; f[i + 8] = row[kb + 16 + i]; }
  return f;
}
// B (32x16 bf16): lane n=lane&15; elems 0..7 -> K=kb..kb+7, elems 8..15 -> K=kb+16..kb+23, kb=(lane<16?0:8); row = weight row n (K contiguous)
__device__ __forceinline__ v16bf ld_b16(const bf16_t* row, int lane) {
  const int kb = (lane & 16) ? 8 : 0;
  v16bf f;
#pragma unroll
  for (int i = 0; i < 8; ++i) { f[i] = row[kb + i]; f[i + 8] = row[kb + 16 + i]; }
  return f;
}
__device__ __forceinline__ v8f wmma_bf16(v16bf a, v16bf b, v8f c) {
  return __builtin_amdgcn_wmma_f32_16x16x32_bf16(false, a, false, b, (short)0, c, false, false);
}
__device__ __forceinline__ v8f wmma_f32(v2f a, v2f b, v8f c) {
  return __builtin_amdgcn_wmma_f32_16x16x4_f32(false, a, false, b, (short)0, c, false, false);
}
// K-loop for C += A(MxK) * W(NxK)^T, K multiple of 32
__device__ __forceinline__ v8f gemm_k(const bf16_t* arow, const bf16_t* wrow, int K, int lane, v8f acc) {
  for (int kk = 0; kk < K; kk += 32)
    acc = wmma_bf16(ld_a16(arow + kk, lane), ld_b16(wrow + kk, lane), acc);
  return acc;
}

// ---------------- elementwise utility kernels ----------------
__global__ void k_fill_f32(float* p, long n, float v) {
  long i = (long)blockIdx.x * blockDim.x + threadIdx.x;
  if (i < n) p[i] = v;
}
__global__ void k_fill_bf16(bf16_t* p, long n, float v) {
  long i = (long)blockIdx.x * blockDim.x + threadIdx.x;
  if (i < n) p[i] = (bf16_t)v;
}
__global__ void k_fill_i32(int* p, long n, int v) {
  long i = (long)blockIdx.x * blockDim.x + threadIdx.x;
  if (i < n) p[i] = v;
}
__global__ void k_cvt_bf16(const float* s, bf16_t* d, long n) {
  long i = (long)blockIdx.x * blockDim.x + threadIdx.x;
  if (i < n) d[i] = (bf16_t)s[i];
}

// ---------------- TTFS encoder: emb lookup + sigmoid -> one-hot over T ----------------
__global__ void k_encode(const int* ids, const float* emb, const float* scaling, bf16_t* h) {
  long i = (long)blockIdx.x * blockDim.x + threadIdx.x;
  if (i >= (long)B_ * S_ * DM_) return;
  int d = (int)(i % DM_);
  long bs = i / DM_;
  int s = (int)(bs % S_), b = (int)(bs / S_);
  float tok = emb[(long)ids[b * S_ + s] * DM_ + d];
  float act = 1.f / (1.f + __expf(-scaling[0] * tok));
  int tt = (int)rintf(MAXLAT_ * (1.f - act));
  tt = min(max(tt, 0), T_ - 1);
  long base = (long)b * T_ * S_ * DM_ + (long)s * DM_ + d;
#pragma unroll
  for (int t2 = 0; t2 < T_; ++t2)
    h[base + (long)t2 * S_ * DM_] = (bf16_t)((t2 == tt) ? 1.f : 0.f);
}

// ---------------- per-step activity flags: any(x_t > 0) ----------------
__global__ void k_flags(const bf16_t* x, int* flags) {
  long i = (long)blockIdx.x * blockDim.x + threadIdx.x;
  if (i >= (long)B_ * S_ * DM_) return;
  long b = i / ((long)S_ * DM_);
  long r = i % ((long)S_ * DM_);
  const bf16_t* p = x + b * (long)T_ * S_ * DM_ + r;
  int lane = threadIdx.x & 31;
#pragma unroll
  for (int t = 0; t < T_; ++t) {
    int sp = ((float)p[(long)t * S_ * DM_] > 0.f) ? 1 : 0;
    if (__ballot(sp)) { if (lane == 0) atomicOr(&flags[t], 1); }
  }
}

// ---------------- SSM state update: su = h@A^T (+ x_t@B^T if active); LIF ----------------
__global__ void k_ssm_state(const bf16_t* hs_in, bf16_t* hs_out, const bf16_t* x, int t,
                            const bf16_t* Ab, const bf16_t* Bb,
                            float* v_s, const float* thr_s, float* cnt_s, const int* flags) {
  int lane = threadIdx.x & 31, wvx = threadIdx.x >> 5;
  int tile = blockIdx.x * 4 + wvx;            // 128 m-tiles x 4 n-tiles = 512
  int tm = tile >> 2, tn = tile & 3;
  int ra = tm * 16 + (lane & 15);
  int n  = tn * 16 + (lane & 15);
  v8f acc = {};
  acc = gemm_k(hs_in + (long)ra * DS_, Ab + (long)n * DS_, DS_, lane, acc);
  if (flags[t]) {
    int b = ra >> 9, s = ra & 511;
    acc = gemm_k(x + ((long)(b * T_ + t) * S_ + s) * DM_, Bb + (long)n * DM_, DM_, lane, acc);
  }
  int half = (lane & 16) ? 8 : 0;
  float thr = thr_s[n];
#pragma unroll
  for (int e = 0; e < 8; ++e) {
    int m = tm * 16 + e + half;
    long idx = (long)m * DS_ + n;
    float vv = v_s[idx] * DECAY_ + acc[e];
    float sp = (vv - thr >= 0.f) ? 1.f : 0.f;
    v_s[idx] = vv * (1.f - sp);
    hs_out[idx] = (bf16_t)sp;
    if (sp != 0.f) atomicAdd(&cnt_s[n], 1.f);
  }
}

// ---------------- SSM output: ou = h2@C^T + x_t@D^T; LIF; write spikes to h_next[:,t] ----
__global__ void k_ssm_out(const bf16_t* h2, const bf16_t* x, bf16_t* hout, int t,
                          const bf16_t* Cb, const bf16_t* Db,
                          float* v_o, const float* thr_o, float* cnt_o, const int* flags) {
  int lane = threadIdx.x & 31, wvx = threadIdx.x >> 5;
  int tile = blockIdx.x * 4 + wvx;            // 128 x 32 = 4096
  int tm = tile >> 5, tn = tile & 31;
  int half = (lane & 16) ? 8 : 0;
  int n = tn * 16 + (lane & 15);
  if (!flags[t]) {                             // inactive: out = zeros, v_o/thr_o untouched
#pragma unroll
    for (int e = 0; e < 8; ++e) {
      int m = tm * 16 + e + half;
      int b = m >> 9, s = m & 511;
      hout[((long)(b * T_ + t) * S_ + s) * DM_ + n] = (bf16_t)0.f;
    }
    return;
  }
  int ra = tm * 16 + (lane & 15);
  int rb = ra >> 9, rs = ra & 511;
  v8f acc = {};
  acc = gemm_k(h2 + (long)ra * DS_, Cb + (long)n * DS_, DS_, lane, acc);
  acc = gemm_k(x + ((long)(rb * T_ + t) * S_ + rs) * DM_, Db + (long)n * DM_, DM_, lane, acc);
  float thr = thr_o[n];
#pragma unroll
  for (int e = 0; e < 8; ++e) {
    int m = tm * 16 + e + half;
    int b = m >> 9, s = m & 511;
    long vidx = (long)m * DM_ + n;
    float vv = v_o[vidx] * DECAY_ + acc[e];
    float sp = (vv - thr >= 0.f) ? 1.f : 0.f;
    v_o[vidx] = vv * (1.f - sp);
    hout[((long)(b * T_ + t) * S_ + s) * DM_ + n] = (bf16_t)sp;
    if (sp != 0.f) atomicAdd(&cnt_o[n], 1.f);
  }
}

// ---------------- adaptive threshold finalize: thr += ADAPT*(mean_spike - 0.1) ----------
__global__ void k_thr(float* thr, float* cnt, int n, const int* flags, int t, int gated) {
  int i = blockIdx.x * blockDim.x + threadIdx.x;
  if (i >= n) return;
  if (gated && !flags[t]) return;
  thr[i] += ADAPT_ * (cnt[i] * (1.f / (B_ * S_)) - 0.1f);
  cnt[i] = 0.f;
}

// ---------------- q/k/v projection: out[(b,s,t),:] = x[b,t,s,:] @ W^T + bias ----------
__global__ void k_proj(const bf16_t* x, const bf16_t* W, const float* bias, float* out) {
  int lane = threadIdx.x & 31, wvx = threadIdx.x >> 5;
  int tile = blockIdx.x * 4 + wvx;            // 2048 x 32 = 65536
  int tm = tile >> 5, tn = tile & 31;
  int r = tm * 16 + (lane & 15);              // row = (b*S+s)*T + t
  int b = r / (S_ * T_);
  int rem = r % (S_ * T_);
  int s = rem / T_, tt = rem % T_;
  const bf16_t* arow = x + ((long)(b * T_ + tt) * S_ + s) * DM_;
  int n = tn * 16 + (lane & 15);
  const bf16_t* wrow = W + (long)n * DM_;
  v8f acc = {};
  for (int kk = 0; kk < DM_; kk += 32) {
    __builtin_prefetch(wrow + kk + 128, 0, 1);
    acc = wmma_bf16(ld_a16(arow + kk, lane), ld_b16(wrow + kk, lane), acc);
  }
  int half = (lane & 16) ? 8 : 0;
  float bvv = bias[n];
#pragma unroll
  for (int e = 0; e < 8; ++e) {
    int m = tm * 16 + e + half;
    out[(long)m * DM_ + n] = acc[e] + bvv;
  }
}

// ---------------- attention core: f32 WMMA 16x16x4 scores + softmax + AV ----------------
__global__ void k_attn(const float* q, const float* k, const float* v, bf16_t* av) {
  __shared__ float wls[4][16][17];
  int lane = threadIdx.x & 31, wvx = threadIdx.x >> 5;
  int gid = blockIdx.x * 4 + wvx;             // (b*S+s)*H + h, total 8192
  int h = gid & 3;
  int bs = gid >> 2;
  const long rowbase = (long)bs * T_;
  const float* qb = q + rowbase * DM_ + h * DH_;
  const float* kb = k + rowbase * DM_ + h * DH_;
  const float* vb = v + rowbase * DM_ + h * DH_;
  int n = lane & 15;
  int half = (lane & 16) ? 8 : 0;
  const int kbo = (lane & 16) ? 2 : 0;
  const float* qrow = qb + (long)n * DM_;     // A row t = n
  const float* krow = kb + (long)n * DM_;     // B col u = n (scores[t,u] = q[t,:].k[u,:])
  v8f sc = {};
#pragma unroll
  for (int kk = 0; kk < DH_; kk += 4) {
    v2f a;  a[0] = qrow[kk + kbo];  a[1] = qrow[kk + kbo + 1];
    v2f bb; bb[0] = krow[kk + kbo]; bb[1] = krow[kk + kbo + 1];
    sc = wmma_f32(a, bb, sc);
  }
  const float scale = 0.08838834764831845f;   // 1/sqrt(128)
  // softmax per row: lanes 0-15 hold u=0..15 for rows e, lanes 16-31 for rows e+8
  float w8[8];
#pragma unroll
  for (int e = 0; e < 8; ++e) {
    float xv = sc[e] * scale;
    float mx = xv;
#pragma unroll
    for (int msk = 8; msk >= 1; msk >>= 1) mx = fmaxf(mx, __shfl_xor(mx, msk, 32));
    float p = __expf(xv - mx);
    float sm = p;
#pragma unroll
    for (int msk = 8; msk >= 1; msk >>= 1) sm += __shfl_xor(sm, msk, 32);
    w8[e] = p / sm;
  }
#pragma unroll
  for (int e = 0; e < 8; ++e) wls[wvx][e + half][n] = w8[e];
  __syncthreads();
  // av[t,d] = sum_u w[t,u] * v[u,d]; 8 n-tiles of 16 over dh=128, K=16 in f32 WMMA steps
#pragma unroll
  for (int nt = 0; nt < 8; ++nt) {
    v8f acc = {};
#pragma unroll
    for (int k0 = 0; k0 < 16; k0 += 4) {
      v2f a;  a[0] = wls[wvx][n][k0 + kbo];       a[1] = wls[wvx][n][k0 + kbo + 1];
      v2f bb; bb[0] = vb[(long)(k0 + kbo) * DM_ + nt * 16 + n];
              bb[1] = vb[(long)(k0 + kbo + 1) * DM_ + nt * 16 + n];
      acc = wmma_f32(a, bb, acc);
    }
#pragma unroll
    for (int e = 0; e < 8; ++e) {
      int t = e + half;
      av[(rowbase + t) * DM_ + h * DH_ + nt * 16 + n] = (bf16_t)acc[e];
    }
  }
}

// ---------------- out-projection + residual: h_final[b,t,s,:] = x + av@Wo^T + bo -------
__global__ void k_outproj(const bf16_t* av, const bf16_t* Wo, const float* bo,
                          const bf16_t* xsp, float* hfin) {
  int lane = threadIdx.x & 31, wvx = threadIdx.x >> 5;
  int tile = blockIdx.x * 4 + wvx;            // 65536
  int tm = tile >> 5, tn = tile & 31;
  int r = tm * 16 + (lane & 15);
  const bf16_t* arow = av + (long)r * DM_;
  int n = tn * 16 + (lane & 15);
  const bf16_t* wrow = Wo + (long)n * DM_;
  v8f acc = {};
  for (int kk = 0; kk < DM_; kk += 32) {
    __builtin_prefetch(wrow + kk + 128, 0, 1);
    acc = wmma_bf16(ld_a16(arow + kk, lane), ld_b16(wrow + kk, lane), acc);
  }
  int half = (lane & 16) ? 8 : 0;
  float bvv = bo[n];
#pragma unroll
  for (int e = 0; e < 8; ++e) {
    int m = tm * 16 + e + half;              // m = (b*S+s)*T + t
    int b = m / (S_ * T_);
    int rem = m % (S_ * T_);
    int s = rem / T_, tt = rem % T_;
    long oidx = ((long)(b * T_ + tt) * S_ + s) * DM_ + n;
    hfin[oidx] = (float)xsp[oidx] + acc[e] + bvv;
  }
}

// ---------------- time integration: hmean = mean_T(h_final) ----------------
__global__ void k_meanT(const float* hfin, bf16_t* hm) {
  long i = (long)blockIdx.x * blockDim.x + threadIdx.x;
  if (i >= (long)B_ * S_ * DM_) return;
  long b = i / ((long)S_ * DM_);
  long r = i % ((long)S_ * DM_);
  const float* p = hfin + b * (long)T_ * S_ * DM_ + r;
  float acc = 0.f;
#pragma unroll
  for (int t = 0; t < T_; ++t) acc += p[(long)t * S_ * DM_];
  hm[i] = (bf16_t)(acc * (1.f / T_));
}

// ---------------- logits GEMM: (2048 x 512) @ (512 x 32000) + bias ----------------
// Each block owns one 16-row A panel, staged into LDS once via the Tensor Data Mover
// (TDM, CDNA5 ISA §7.3/§8: 2D D#, data_size=2B, tile 512x16). All 4 waves then reuse
// the panel across 16 N-tiles (4 per wave, register-blocked: 1 A-frag feeds 4 WMMAs).
#define NTG_ 16                         // n-tiles per block
__global__ void k_logits(const bf16_t* hm, const bf16_t* Wout, const float* bout, float* out) {
  __shared__ __align__(128) bf16_t atile[16 * DM_];   // 16KB A panel
  int lane = threadIdx.x & 31, wvx = threadIdx.x >> 5;
  const int ngroups = (V_ / 16) / NTG_;               // 2000/16 = 125
  int tm = blockIdx.x / ngroups;                      // 0..127
  int ng = blockIdx.x % ngroups;

  if (threadIdx.x < 32) {                             // wave 0 issues the tensor DMA
    unsigned long long ga = (unsigned long long)(const void*)(hm + (long)tm * 16 * DM_);
    unsigned int lds = (unsigned int)(unsigned long long)(void*)atile; // LDS byte offset
    u32x4 g0;
    g0[0] = 1u;                                       // count=1, user descriptor
    g0[1] = lds;                                      // lds_addr
    g0[2] = (unsigned int)(ga & 0xffffffffu);         // global_addr[31:0]
    g0[3] = (unsigned int)((ga >> 32) & 0x01ffffffu) | (2u << 30); // addr[56:32], type=2
    i32x8 g1;
    g1[0] = (1 << 16);                                // data_size=1 (2 bytes)
    g1[1] = (int)((DM_ & 0xffff) << 16);              // tensor_dim0 = 512 (lo16)
    g1[2] = (16 << 16);                               // dim0 hi=0 | tensor_dim1 = 16
    g1[3] = (int)((DM_ & 0xffff) << 16);              // dim1 hi=0 | tile_dim0 = 512
    g1[4] = 16;                                       // tile_dim1 = 16, tile_dim2 = 0
    g1[5] = DM_;                                      // tensor_dim0_stride = 512 (lo32)
    g1[6] = 0;                                        // stride0 hi | dim1_stride lo
    g1[7] = 0;
    i32x4 gz  = {0, 0, 0, 0};                         // groups 2/3 unused (2D tensor)
    i32x8 gz8 = {0, 0, 0, 0, 0, 0, 0, 0};             // trailing group (clang-23 6-arg form)
    __builtin_amdgcn_tensor_load_to_lds(g0, g1, gz, gz, gz8, 0);
    __builtin_amdgcn_s_wait_tensorcnt(0);
  }
  __syncthreads();                                    // publish A panel to all waves

  int n0 = (ng * NTG_ + wvx * 4) * 16;                // first output column of this wave
  const bf16_t* arow = atile + (long)(lane & 15) * DM_;
  const bf16_t* wrow = Wout + (long)(n0 + (lane & 15)) * DM_;
  v8f acc[4] = {{}, {}, {}, {}};
  for (int kk = 0; kk < DM_; kk += 32) {
    v16bf af = ld_a16(arow + kk, lane);               // one LDS A-frag ...
    __builtin_prefetch(wrow + kk + 128, 0, 1);
#pragma unroll
    for (int j = 0; j < 4; ++j)                       // ... feeds 4 WMMAs
      acc[j] = wmma_bf16(af, ld_b16(wrow + (long)j * 16 * DM_ + kk, lane), acc[j]);
  }
  int half = (lane & 16) ? 8 : 0;
#pragma unroll
  for (int j = 0; j < 4; ++j) {
    int n = n0 + j * 16 + (lane & 15);
    float bvv = bout[n];
#pragma unroll
    for (int e = 0; e < 8; ++e) {
      int m = tm * 16 + e + half;
      out[(long)m * V_ + n] = acc[j][e] + bvv;
    }
  }
}

// ================= host driver =================
static inline long cdiv(long a, long b) { return (a + b - 1) / b; }

extern "C" void kernel_launch(void* const* d_in, const int* in_sizes, int n_in,
                              void* d_out, int out_size, void* d_ws, size_t ws_size,
                              hipStream_t stream) {
  const int*   ids     = (const int*)d_in[0];
  const float* emb     = (const float*)d_in[1];
  const float* scaling = (const float*)d_in[2];
  const float* A_f     = (const float*)d_in[3];
  const float* B_f     = (const float*)d_in[4];
  const float* C_f     = (const float*)d_in[5];
  const float* D_f     = (const float*)d_in[6];
  const float* wq = (const float*)d_in[7];   const float* bq = (const float*)d_in[8];
  const float* wk = (const float*)d_in[9];   const float* bk = (const float*)d_in[10];
  const float* wv = (const float*)d_in[11];  const float* bv = (const float*)d_in[12];
  const float* wo = (const float*)d_in[13];  const float* bo = (const float*)d_in[14];
  const float* wout = (const float*)d_in[15]; const float* bout = (const float*)d_in[16];

  char* wsb = (char*)d_ws;
  size_t off = 0;
  auto alloc = [&](size_t bytes) -> void* {
    off = (off + 255) & ~(size_t)255;
    void* p = wsb + off;
    off += bytes;
    return p;
  };

  const long NH = (long)B_ * T_ * S_ * DM_;      // 16,777,216
  bf16_t* hb0 = (bf16_t*)alloc(NH * 2);
  bf16_t* hb1 = (bf16_t*)alloc(NH * 2);
  float*  qB  = (float*)alloc(NH * 4);
  float*  kB  = (float*)alloc(NH * 4);
  float*  vB  = (float*)alloc(NH * 4);
  bf16_t* avB = (bf16_t*)alloc(NH * 2);
  bf16_t* hm  = (bf16_t*)alloc((long)B_ * S_ * DM_ * 2);
  bf16_t* Ab  = (bf16_t*)alloc(2L * DS_ * DS_ * 2);
  bf16_t* Bb  = (bf16_t*)alloc(2L * DS_ * DM_ * 2);
  bf16_t* Cb  = (bf16_t*)alloc(2L * DM_ * DS_ * 2);
  bf16_t* Db  = (bf16_t*)alloc(2L * DM_ * DM_ * 2);
  bf16_t* wqb = (bf16_t*)alloc((long)DM_ * DM_ * 2);
  bf16_t* wkb = (bf16_t*)alloc((long)DM_ * DM_ * 2);
  bf16_t* wvb = (bf16_t*)alloc((long)DM_ * DM_ * 2);
  bf16_t* wob = (bf16_t*)alloc((long)DM_ * DM_ * 2);
  bf16_t* woutb = (bf16_t*)alloc((long)V_ * DM_ * 2);
  bf16_t* hs0 = (bf16_t*)alloc((long)B_ * S_ * DS_ * 2);
  bf16_t* hs1 = (bf16_t*)alloc((long)B_ * S_ * DS_ * 2);
  float*  v_s = (float*)alloc((long)B_ * S_ * DS_ * 4);
  float*  v_o = (float*)alloc((long)B_ * S_ * DM_ * 4);
  float*  thr_s = (float*)alloc(DS_ * 4);
  float*  thr_o = (float*)alloc(DM_ * 4);
  float*  cnt_s = (float*)alloc(DS_ * 4);
  float*  cnt_o = (float*)alloc(DM_ * 4);
  int*    flags = (int*)alloc(T_ * 4);

  float* logits = (float*)d_out;
  float* hfin   = logits + (long)B_ * S_ * V_;   // second output (h) region

  // weights -> bf16 (binary spikes make the activation side exact in bf16)
  auto cvt = [&](const float* s, bf16_t* d, long n) {
    k_cvt_bf16<<<cdiv(n, 256), 256, 0, stream>>>(s, d, n);
  };
  cvt(A_f, Ab, 2L * DS_ * DS_);
  cvt(B_f, Bb, 2L * DS_ * DM_);
  cvt(C_f, Cb, 2L * DM_ * DS_);
  cvt(D_f, Db, 2L * DM_ * DM_);
  cvt(wq, wqb, (long)DM_ * DM_);
  cvt(wk, wkb, (long)DM_ * DM_);
  cvt(wv, wvb, (long)DM_ * DM_);
  cvt(wo, wob, (long)DM_ * DM_);
  cvt(wout, woutb, (long)V_ * DM_);

  // TTFS encode -> hb0
  k_encode<<<cdiv((long)B_ * S_ * DM_, 256), 256, 0, stream>>>(ids, emb, scaling, hb0);

  bf16_t* xcur = hb0;
  bf16_t* xnxt = hb1;
  for (int l = 0; l < 2; ++l) {
    // per-step activity flags over layer input
    k_fill_i32<<<1, 64, 0, stream>>>(flags, T_, 0);
    k_flags<<<cdiv((long)B_ * S_ * DM_, 256), 256, 0, stream>>>(xcur, flags);
    // reset neuron state
    k_fill_f32<<<cdiv((long)B_ * S_ * DS_, 256), 256, 0, stream>>>(v_s, (long)B_ * S_ * DS_, 0.f);
    k_fill_f32<<<cdiv((long)B_ * S_ * DM_, 256), 256, 0, stream>>>(v_o, (long)B_ * S_ * DM_, 0.f);
    k_fill_bf16<<<cdiv((long)B_ * S_ * DS_, 256), 256, 0, stream>>>(hs0, (long)B_ * S_ * DS_, 0.f);
    k_fill_f32<<<1, 64, 0, stream>>>(thr_s, DS_, 1.f);
    k_fill_f32<<<1, 64, 0, stream>>>(cnt_s, DS_, 0.f);
    k_fill_f32<<<2, 256, 0, stream>>>(thr_o, DM_, 1.f);
    k_fill_f32<<<2, 256, 0, stream>>>(cnt_o, DM_, 0.f);

    const bf16_t* Al = Ab + (long)l * DS_ * DS_;
    const bf16_t* Bl = Bb + (long)l * DS_ * DM_;
    const bf16_t* Cl = Cb + (long)l * DM_ * DS_;
    const bf16_t* Dl = Db + (long)l * DM_ * DM_;

    bf16_t* hin = hs0;
    bf16_t* hout2 = hs1;
    for (int t = 0; t < T_; ++t) {
      k_ssm_state<<<128, 128, 0, stream>>>(hin, hout2, xcur, t, Al, Bl, v_s, thr_s, cnt_s, flags);
      k_thr<<<1, 64, 0, stream>>>(thr_s, cnt_s, DS_, flags, t, 0);
      k_ssm_out<<<1024, 128, 0, stream>>>(hout2, xcur, xnxt, t, Cl, Dl, v_o, thr_o, cnt_o, flags);
      k_thr<<<2, 256, 0, stream>>>(thr_o, cnt_o, DM_, flags, t, 1);
      bf16_t* tmp = hin; hin = hout2; hout2 = tmp;
    }
    bf16_t* tmp = xcur; xcur = xnxt; xnxt = tmp;
  }

  // attention: projections (bf16 WMMA, spike activations exact), core (f32 WMMA), out-proj
  k_proj<<<16384, 128, 0, stream>>>(xcur, wqb, bq, qB);
  k_proj<<<16384, 128, 0, stream>>>(xcur, wkb, bk, kB);
  k_proj<<<16384, 128, 0, stream>>>(xcur, wvb, bv, vB);
  k_attn<<<2048, 128, 0, stream>>>(qB, kB, vB, avB);
  k_outproj<<<16384, 128, 0, stream>>>(avB, wob, bo, xcur, hfin);

  // integrate over time and project to vocab
  k_meanT<<<cdiv((long)B_ * S_ * DM_, 256), 256, 0, stream>>>(hfin, hm);
  k_logits<<<128 * 125, 128, 0, stream>>>(hm, woutb, bout, logits);
}
